// VariationalEncoder_22333829939968
// MI455X (gfx1250) — matrix operand order repaired and checked by reading, not compiled, not run on hardware
//
#include <hip/hip_runtime.h>
#include <hip/hip_bf16.h>
#include <stdint.h>

// ---------- WMMA types ----------
typedef __attribute__((ext_vector_type(16))) __bf16 v16bf;
typedef __attribute__((ext_vector_type(16))) float  v16f;
typedef __attribute__((ext_vector_type(8)))  float  v8f;

union ABfrag { uint32_t u[8]; v16bf v; };
union BFPair { uint32_t u; __bf16 h[2]; };

__device__ __forceinline__ uint32_t pack2bf(float a, float b) {
  BFPair p; p.h[0] = (__bf16)a; p.h[1] = (__bf16)b;   // native RNE converts
  return p.u;
}
__device__ __forceinline__ float4 relu4(float4 v) {
  v.x = fmaxf(v.x, 0.0f); v.y = fmaxf(v.y, 0.0f);
  v.z = fmaxf(v.z, 0.0f); v.w = fmaxf(v.w, 0.0f);
  return v;
}

// ---------------------------------------------------------------------------
// GEMM: C[N x DOUT] = A[N x DIN] @ W[DIN x DOUT], bf16 WMMA, fp32 accum.
// Block = 256 threads = 8 waves; block tile 32 rows x DOUT cols.
// W K-slices staged in LDS pre-swizzled into B-fragment layout:
//   per 16-col tile: sW[(tile*32 + lane)*8 + v] = bf16 pair (K=2v, 2v+1)
//   with lane = col%16 + 16*(K>=16); each lane reads its whole fragment
//   with two conflict-free ds_load_b128.
// ---------------------------------------------------------------------------
template<int DIN, int DOUT>
__global__ __launch_bounds__(256) void gemm_bf16_wmma(
    const float* __restrict__ A, const float* __restrict__ W,
    float* __restrict__ C, int N, int reluA)
{
  constexpr int TILES = DOUT / 64;              // 16-wide col tiles per wave
  constexpr int NT    = DOUT / 16;              // total col tiles in block
  __shared__ uint32_t sW[NT * 32 * 8];          // fragment-major bf16 K-slice

  const int tid  = threadIdx.x;
  const int lane = tid & 31;
  const int wave = tid >> 5;
  const int rowTile = wave & 1;
  const int colGrp  = wave >> 1;                // 0..3
  const int rowBase = blockIdx.x * 32 + rowTile * 16;
  const int tileBase = colGrp * TILES;          // first 16-col tile of this wave
  const int colBase  = tileBase * 16;

  const int m  = lane & 15;
  const int hi = lane >> 4;
  const int arow = (rowBase + m < N) ? (rowBase + m) : (N - 1);

  v8f acc[TILES] = {};

  for (int k0 = 0; k0 < DIN; k0 += 32) {
    __syncthreads();
    // cooperative swizzled load of W[k0..k0+32)[0..DOUT) into fragment layout
    #pragma unroll
    for (int i = tid; i < 16 * DOUT; i += 256) {
      const int kk2 = i / DOUT;                 // K pair index 0..15
      const int ff  = i % DOUT;                 // column (coalesced)
      const int k   = kk2 * 2;
      const float w0 = W[(size_t)(k0 + k    ) * DOUT + ff];
      const float w1 = W[(size_t)(k0 + k + 1) * DOUT + ff];
      const int lanepos = (ff & 15) + ((k >> 4) << 4);
      sW[(((ff >> 4) * 32 + lanepos) << 3) + ((k & 15) >> 1)] = pack2bf(w0, w1);
    }
    __syncthreads();

    // A fragment: 16x32 bf16; lanes 0-15 K {0..15}, lanes 16-31 K {16..31}
    ABfrag af;
    {
      const float4* ap = (const float4*)(A + (size_t)arow * DIN + k0 + hi * 16);
      float4 r0 = ap[0], r1 = ap[1], r2 = ap[2], r3 = ap[3];
      if (reluA) { r0 = relu4(r0); r1 = relu4(r1); r2 = relu4(r2); r3 = relu4(r3); }
      v16f fa;
      fa[0]  = r0.x; fa[1]  = r0.y; fa[2]  = r0.z; fa[3]  = r0.w;
      fa[4]  = r1.x; fa[5]  = r1.y; fa[6]  = r1.z; fa[7]  = r1.w;
      fa[8]  = r2.x; fa[9]  = r2.y; fa[10] = r2.z; fa[11] = r2.w;
      fa[12] = r3.x; fa[13] = r3.y; fa[14] = r3.z; fa[15] = r3.w;
      af.v = __builtin_convertvector(fa, v16bf);  // packed native cvt
    }

    // B fragments: two ds_load_b128 per tile, then WMMA
    #pragma unroll
    for (int t = 0; t < TILES; ++t) {
      const int tt = tileBase + t;
      const uint4* pp = (const uint4*)&sW[(tt * 32 + lane) << 3];
      ABfrag bfrag;
      *(uint4*)&bfrag.u[0] = pp[0];
      *(uint4*)&bfrag.u[4] = pp[1];
      acc[t] = __builtin_amdgcn_wmma_f32_16x16x32_bf16(
          false, af.v, false, bfrag.v, (short)0, acc[t], false, false);
    }
  }

  // store: VGPR r holds row M=r (lanes 0-15) / M=r+8 (lanes 16-31), N = lane&15
  #pragma unroll
  for (int t = 0; t < TILES; ++t) {
    const int n = colBase + t * 16 + m;
    #pragma unroll
    for (int r = 0; r < 8; ++r) {
      const int row = rowBase + r + hi * 8;
      if (row < N) C[(size_t)row * DOUT + n] = acc[t][r];
    }
  }
}

// ---------------------------------------------------------------------------
// GCN normalization helpers
// ---------------------------------------------------------------------------
__global__ void deg_init(float* __restrict__ deg, int N) {
  int i = blockIdx.x * blockDim.x + threadIdx.x;
  if (i < N) deg[i] = 1.0f;                       // self-loop contribution
}
__global__ void deg_count(const int* __restrict__ col, float* __restrict__ deg, int E) {
  int e = blockIdx.x * blockDim.x + threadIdx.x;
  if (e < E) atomicAdd(&deg[col[e]], 1.0f);
}
__global__ void deg_rsqrt(float* __restrict__ deg, int N) {
  int i = blockIdx.x * blockDim.x + threadIdx.x;
  if (i < N) deg[i] = rsqrtf(deg[i]);             // deg >= 1 always
}
__global__ void edge_norm(const int* __restrict__ row, const int* __restrict__ col,
                          const float* __restrict__ dinv, float* __restrict__ nrm, int E) {
  int e = blockIdx.x * blockDim.x + threadIdx.x;
  if (e < E) nrm[e] = dinv[row[e]] * dinv[col[e]];
}

// ---------------------------------------------------------------------------
// Aggregation: out = bias + dinv^2 * h (fused init+self-loop), then edge scatter
// ---------------------------------------------------------------------------
template<int DOUT>
__global__ void init_self(const float* __restrict__ h, const float* __restrict__ dinv,
                          const float* __restrict__ b, float* __restrict__ out, int N) {
  constexpr int CH = DOUT / 4;
  int i = blockIdx.x * blockDim.x + threadIdx.x;
  int node = i / CH;
  if (node < N) {
    int f = (i % CH) * 4;
    float s = dinv[node]; s *= s;
    const float4 v  = *(const float4*)(h + (size_t)node * DOUT + f);
    const float4 bb = *(const float4*)(b + f);
    float4 o;
    o.x = bb.x + v.x * s; o.y = bb.y + v.y * s;
    o.z = bb.z + v.z * s; o.w = bb.w + v.w * s;
    *(float4*)(out + (size_t)node * DOUT + f) = o;
  }
}

template<int DOUT>
__global__ void edge_scatter(const int* __restrict__ row, const int* __restrict__ col,
                             const float* __restrict__ nrm, const float* __restrict__ h,
                             float* __restrict__ out, int E) {
  constexpr int CH = DOUT / 4;
  int i = blockIdx.x * blockDim.x + threadIdx.x;
  int e = i / CH;
  if (e < E) {
    int f = (i % CH) * 4;
    const int r = row[e], c = col[e];
    const float s = nrm[e];
    const float4 v = *(const float4*)(h + (size_t)r * DOUT + f);
    float* o = out + (size_t)c * DOUT + f;
    atomicAdd(o + 0, v.x * s);
    atomicAdd(o + 1, v.y * s);
    atomicAdd(o + 2, v.z * s);
    atomicAdd(o + 3, v.w * s);
  }
}

// ---------------------------------------------------------------------------
// Host-side helpers / launch
// ---------------------------------------------------------------------------
template<int DOUT>
static void aggregate(hipStream_t s, const int* row, const int* col, const float* nrm,
                      const float* dinv, const float* h, const float* bias,
                      float* out, int N, int E) {
  const int TB = 256;
  long long sl = (long long)N * (DOUT / 4);
  init_self<DOUT><<<(unsigned)((sl + TB - 1) / TB), TB, 0, s>>>(h, dinv, bias, out, N);
  long long ec = (long long)E * (DOUT / 4);
  edge_scatter<DOUT><<<(unsigned)((ec + TB - 1) / TB), TB, 0, s>>>(row, col, nrm, h, out, E);
}

extern "C" void kernel_launch(void* const* d_in, const int* in_sizes, int n_in,
                              void* d_out, int out_size, void* d_ws, size_t ws_size,
                              hipStream_t stream) {
  (void)n_in; (void)out_size; (void)ws_size;
  const int FIN = 256, HH = 256, FOUT = 128;

  const float* x  = (const float*)d_in[0];
  const int*   ei = (const int*)d_in[1];
  const int N = in_sizes[0] / FIN;
  const int E = in_sizes[1] / 2;
  const int* rowI = ei;
  const int* colI = ei + E;

  const float* Wt[2][3] = {
    { (const float*)d_in[2], (const float*)d_in[4],  (const float*)d_in[6]  },
    { (const float*)d_in[8], (const float*)d_in[10], (const float*)d_in[12] } };
  const float* Bs[2][3] = {
    { (const float*)d_in[3], (const float*)d_in[5],  (const float*)d_in[7]  },
    { (const float*)d_in[9], (const float*)d_in[11], (const float*)d_in[13] } };

  auto al = [](size_t v) { return (v + 255) & ~(size_t)255; };
  char* p = (char*)d_ws;
  float* dinv = (float*)p; p += al((size_t)N * 4);
  float* nrm  = (float*)p; p += al((size_t)E * 4);
  float* tA   = (float*)p; p += al((size_t)N * HH * 4);
  float* tB   = (float*)p; p += al((size_t)N * HH * 4);

  const int TB = 256;
  deg_init <<<(N + TB - 1) / TB, TB, 0, stream>>>(dinv, N);
  deg_count<<<(E + TB - 1) / TB, TB, 0, stream>>>(colI, dinv, E);
  deg_rsqrt<<<(N + TB - 1) / TB, TB, 0, stream>>>(dinv, N);
  edge_norm<<<(E + TB - 1) / TB, TB, 0, stream>>>(rowI, colI, dinv, nrm, E);

  dim3 ggrid((N + 31) / 32);
  float* outF = (float*)d_out;

  for (int br = 0; br < 2; ++br) {
    // layer 0: x -> tA, aggregate -> tB
    gemm_bf16_wmma<256, 256><<<ggrid, 256, 0, stream>>>(x, Wt[br][0], tA, N, 0);
    aggregate<256>(stream, rowI, colI, nrm, dinv, tA, Bs[br][0], tB, N, E);
    // layer 1 (ReLU fused into A conversion)
    gemm_bf16_wmma<256, 256><<<ggrid, 256, 0, stream>>>(tB, Wt[br][1], tA, N, 1);
    aggregate<256>(stream, rowI, colI, nrm, dinv, tA, Bs[br][1], tB, N, E);
    // layer 2: 256 -> 128, final aggregate directly into d_out
    gemm_bf16_wmma<256, 128><<<ggrid, 256, 0, stream>>>(tB, Wt[br][2], tA, N, 1);
    aggregate<128>(stream, rowI, colI, nrm, dinv, tA, Bs[br][2],
                   outF + (size_t)br * N * FOUT, N, E);
  }
}
